// MultiHeadAttention_89180700934768
// MI455X (gfx1250) — matrix operand order repaired and checked by reading, not compiled, not run on hardware
//
#include <hip/hip_runtime.h>

#define B_ 2
#define S_ 2048
#define D_ 512
#define H_ 8
#define U_ 64
#define WIN 128

typedef __attribute__((ext_vector_type(16))) __bf16 v16bf;
typedef __attribute__((ext_vector_type(8)))  __bf16 v8bf;
typedef __attribute__((ext_vector_type(8)))  float  v8f;
typedef __attribute__((ext_vector_type(8)))  float  f32x8;

union V16 { v16bf v; v8bf h[2]; };

// ---------------------------------------------------------------------------
// Kernel 1: transpose + convert weights to bf16: Wt[sel][n][k] = W[sel][k][n]
// ---------------------------------------------------------------------------
__global__ __launch_bounds__(256) void wt_kernel(const float* __restrict__ Wq,
                                                 const float* __restrict__ Wk,
                                                 const float* __restrict__ Wv,
                                                 __bf16* __restrict__ Wt) {
  int idx = blockIdx.x * blockDim.x + threadIdx.x;   // 3 * 512 * 512 threads
  int sel = idx >> 18;
  int rem = idx & 262143;
  int n = rem >> 9, k = rem & 511;
  const float* W = (sel == 0) ? Wq : ((sel == 1) ? Wk : Wv);
  Wt[(size_t)sel * 262144 + (size_t)n * 512 + k] = (__bf16)W[(size_t)k * 512 + n];
}

// ---------------------------------------------------------------------------
// Kernel 2: projections Y = X @ W + b  via bf16 WMMA, f32 accumulate.
// Q,K -> row-major bf16 [B*S][512]; V -> transposed bf16 [B][H][U][S]
// One wave computes a 16-row x 64-col (one head) tile.
// ---------------------------------------------------------------------------
__global__ __launch_bounds__(256) void proj_kernel(const float* __restrict__ Xq,
                                                   const float* __restrict__ Xk,
                                                   const float* __restrict__ Xv,
                                                   const __bf16* __restrict__ Wt,
                                                   const float* __restrict__ bq,
                                                   const float* __restrict__ bk,
                                                   const float* __restrict__ bv,
                                                   __bf16* __restrict__ Qb,
                                                   __bf16* __restrict__ Kb,
                                                   __bf16* __restrict__ Vt) {
  int wave   = (blockIdx.x * blockDim.x + threadIdx.x) >> 5;
  int lane   = threadIdx.x & 31;
  int lane16 = lane & 15;
  int hf     = lane >> 4;          // half-wave select

  int sel = wave >> 11;            // 0:Q 1:K 2:V   (2048 waves each)
  int rem = wave & 2047;
  int rt  = rem >> 3;              // row tile 0..255 (16 rows each over B*S)
  int ct  = rem & 7;               // head / 64-col tile

  const float*  X    = (sel == 0) ? Xq : ((sel == 1) ? Xk : Xv);
  const __bf16* W    = Wt + (size_t)sel * 262144;
  const float*  bias = (sel == 0) ? bq : ((sel == 1) ? bk : bv);

  int arow = rt * 16 + lane16;                 // A-fragment row for this lane
  const float* xrow = X + (size_t)arow * D_;

  v8f acc[4] = {};

  for (int kc = 0; kc < 16; ++kc) {
    int k0 = kc * 32;
    // A fragment: lane holds K = {k0+8hf..+7} then {k0+16+8hf..+7}
    f32x8 xa = *(const f32x8*)(xrow + k0 + 8 * hf);
    f32x8 xb = *(const f32x8*)(xrow + k0 + 16 + 8 * hf);
    V16 a;
#pragma unroll
    for (int i = 0; i < 8; ++i) {
      a.h[0][i] = (__bf16)xa[i];
      a.h[1][i] = (__bf16)xb[i];
    }
#pragma unroll
    for (int nc = 0; nc < 4; ++nc) {
      int n = ct * 64 + nc * 16 + lane16;      // B column for this lane
      V16 bu;
      bu.h[0] = *(const v8bf*)(W + (size_t)n * 512 + k0 + 8 * hf);
      bu.h[1] = *(const v8bf*)(W + (size_t)n * 512 + k0 + 16 + 8 * hf);
      v16bf bf = bu.v;
      acc[nc] = __builtin_amdgcn_wmma_f32_16x16x32_bf16(
          false, a.v, false, bf, (short)0, acc[nc], false, false);
    }
  }

#pragma unroll
  for (int nc = 0; nc < 4; ++nc) {
    int col = ct * 64 + nc * 16 + lane16;
    float bvv = bias[col];
#pragma unroll
    for (int r = 0; r < 8; ++r) {
      int grow = rt * 16 + r + 8 * hf;         // D-tile row
      float y = acc[nc][r] + bvv;
      if (sel == 0) {
        Qb[(size_t)grow * 512 + col] = (__bf16)y;
      } else if (sel == 1) {
        Kb[(size_t)grow * 512 + col] = (__bf16)y;
      } else {
        int bb = grow >> 11;                   // batch
        int s  = grow & 2047;                  // seq pos
        int u  = nc * 16 + lane16;             // unit within head ct
        Vt[(((size_t)bb * H_ + ct) * U_ + u) * S_ + s] = (__bf16)y;
      }
    }
  }
}

// ---------------------------------------------------------------------------
// Kernel 3: banded flash attention. One wave per (b, h, 16-query tile).
// 9 iterations x 32 keys cover [q0-128, q0+159] (extras masked).
// ---------------------------------------------------------------------------
__global__ __launch_bounds__(256) void attn_kernel(const __bf16* __restrict__ Qb,
                                                   const __bf16* __restrict__ Kb,
                                                   const __bf16* __restrict__ Vt,
                                                   float* __restrict__ out) {
  __shared__ __align__(64) __bf16 pst[8][16 * 32];   // P staging per wave

  int wl     = threadIdx.x >> 5;
  int wave   = blockIdx.x * 8 + wl;
  int lane   = threadIdx.x & 31;
  int lane16 = lane & 15;
  int hf     = lane >> 4;

  int b   = wave >> 10;            // / (H * S/16)
  int rem = wave & 1023;
  int h   = rem >> 7;
  int q0  = (rem & 127) * 16;

  // Q A-fragments (two 32-wide U chunks)
  const __bf16* qrow = Qb + (size_t)(b * S_ + q0 + lane16) * 512 + h * 64;
  V16 qa[2];
#pragma unroll
  for (int uc = 0; uc < 2; ++uc) {
    qa[uc].h[0] = *(const v8bf*)(qrow + uc * 32 + 8 * hf);
    qa[uc].h[1] = *(const v8bf*)(qrow + uc * 32 + 16 + 8 * hf);
  }

  float m[8], l[8];
  v8f O[4] = {};
#pragma unroll
  for (int r = 0; r < 8; ++r) { m[r] = -1e30f; l[r] = 0.0f; }

  __bf16* myp = pst[wl];
  const __bf16* vbase = Vt + ((size_t)b * H_ + h) * U_ * S_;

  for (int it = 0; it < 9; ++it) {
    int kb = q0 - WIN + it * 32;

    // ---- S = (Q K^T) / 8, two 16-key tiles ----
    v8f st[2];
#pragma unroll
    for (int t = 0; t < 2; ++t) {
      int key  = kb + 16 * t + lane16;
      int keyc = key < 0 ? 0 : (key > S_ - 1 ? S_ - 1 : key);
      const __bf16* krow = Kb + (size_t)(b * S_ + keyc) * 512 + h * 64;
      v8f s = {};
#pragma unroll
      for (int uc = 0; uc < 2; ++uc) {
        V16 ku;
        ku.h[0] = *(const v8bf*)(krow + uc * 32 + 8 * hf);
        ku.h[1] = *(const v8bf*)(krow + uc * 32 + 16 + 8 * hf);
        v16bf kf = ku.v;
        s = __builtin_amdgcn_wmma_f32_16x16x32_bf16(
            false, qa[uc].v, false, kf, (short)0, s, false, false);
      }
#pragma unroll
      for (int r = 0; r < 8; ++r) {
        int q = q0 + r + 8 * hf;
        bool in = (key >= q - WIN) && (key <= q + WIN) && (key >= 0) && (key < S_);
        s[r] = in ? s[r] * 0.125f : -1e30f;
      }
      st[t] = s;
    }

    // ---- online softmax: row max across 16-lane group ----
    float tmax[8];
#pragma unroll
    for (int r = 0; r < 8; ++r) tmax[r] = fmaxf(st[0][r], st[1][r]);
#pragma unroll
    for (int msk = 1; msk < 16; msk <<= 1) {
#pragma unroll
      for (int r = 0; r < 8; ++r)
        tmax[r] = fmaxf(tmax[r], __shfl_xor(tmax[r], msk, 32));
    }

    float alpha[8], psum[8];
#pragma unroll
    for (int r = 0; r < 8; ++r) {
      float mn = fmaxf(m[r], tmax[r]);
      alpha[r] = __expf(m[r] - mn);
      m[r] = mn;
      l[r] *= alpha[r];
      psum[r] = 0.0f;
    }
#pragma unroll
    for (int nc = 0; nc < 4; ++nc) {
#pragma unroll
      for (int r = 0; r < 8; ++r) O[nc][r] *= alpha[r];
    }

    // ---- P = exp(S - m), stage through LDS to build A-fragment ----
#pragma unroll
    for (int t = 0; t < 2; ++t) {
#pragma unroll
      for (int r = 0; r < 8; ++r) {
        float p = __expf(st[t][r] - m[r]);
        psum[r] += p;
        myp[(r + 8 * hf) * 32 + t * 16 + lane16] = (__bf16)p;
      }
    }
#pragma unroll
    for (int msk = 1; msk < 16; msk <<= 1) {
#pragma unroll
      for (int r = 0; r < 8; ++r) psum[r] += __shfl_xor(psum[r], msk, 32);
    }
#pragma unroll
    for (int r = 0; r < 8; ++r) l[r] += psum[r];

    V16 pa;   // A-fragment of P (16x32 over keys)
    pa.h[0] = *(const v8bf*)(myp + lane16 * 32 + 16 * hf);
    pa.h[1] = *(const v8bf*)(myp + lane16 * 32 + 16 * hf + 8);

    // ---- O += P V : V B-fragment = 16 consecutive keys of Vt row ----
    int kvb = kb + 16 * hf;
    kvb = kvb < 0 ? 0 : (kvb > S_ - 16 ? S_ - 16 : kvb);
#pragma unroll
    for (int nc = 0; nc < 4; ++nc) {
      int u = nc * 16 + lane16;
      v16bf vf = *(const v16bf*)(vbase + (size_t)u * S_ + kvb);
      O[nc] = __builtin_amdgcn_wmma_f32_16x16x32_bf16(
          false, pa.v, false, vf, (short)0, O[nc], false, false);
    }
  }

  // ---- normalize and write fp32 output [B][S][H*U] ----
#pragma unroll
  for (int nc = 0; nc < 4; ++nc) {
    int col = h * 64 + nc * 16 + lane16;
#pragma unroll
    for (int r = 0; r < 8; ++r) {
      int srow = q0 + r + 8 * hf;
      out[(size_t)(b * S_ + srow) * 512 + col] = O[nc][r] / l[r];
    }
  }
}

// ---------------------------------------------------------------------------
extern "C" void kernel_launch(void* const* d_in, const int* in_sizes, int n_in,
                              void* d_out, int out_size, void* d_ws, size_t ws_size,
                              hipStream_t stream) {
  (void)in_sizes; (void)n_in; (void)out_size; (void)ws_size;
  const float* query = (const float*)d_in[0];
  const float* key_  = (const float*)d_in[1];
  const float* value = (const float*)d_in[2];
  const float* Wq    = (const float*)d_in[3];
  const float* Wk    = (const float*)d_in[4];
  const float* Wv    = (const float*)d_in[5];
  const float* bq    = (const float*)d_in[6];
  const float* bk    = (const float*)d_in[7];
  const float* bv    = (const float*)d_in[8];
  float* out = (float*)d_out;

  char* ws = (char*)d_ws;
  __bf16* Wt = (__bf16*)ws;                              // 3 * 512*512 bf16 = 1.5 MB
  __bf16* Qb = (__bf16*)(ws + (size_t)3 * 262144 * 2);   // 2 MiElem bf16 = 4 MB
  __bf16* Kb = Qb + (size_t)2097152;
  __bf16* Vt = Kb + (size_t)2097152;

  wt_kernel<<<3072, 256, 0, stream>>>(Wq, Wk, Wv, Wt);
  proj_kernel<<<768, 256, 0, stream>>>(query, key_, value, Wt, bq, bk, bv, Qb, Kb, Vt);
  attn_kernel<<<256, 256, 0, stream>>>(Qb, Kb, Vt, out);
}
